// HybridLlamaAttention_55027120996633
// MI455X (gfx1250) — matrix operand order repaired and checked by reading, not compiled, not run on hardware
//
#include <hip/hip_runtime.h>
#include <hip/hip_bf16.h>

// ---------------------------------------------------------------------------
// Gemma2-style attention layer for gfx1250 (MI455X).
// All matmuls on v_wmma_f32_16x16x32_f16; flash-attention K/V tiles staged
// into LDS via CDNA5 async-to-LDS DMA (ASYNCcnt), double buffered.
// B=1, S=2048, D=2048, H=16, DH=128.
// ---------------------------------------------------------------------------

typedef _Float16 v8h  __attribute__((ext_vector_type(8)));
typedef _Float16 v16h __attribute__((ext_vector_type(16)));
typedef float    v8f  __attribute__((ext_vector_type(8)));

#define S_LEN   2048
#define DMODEL  2048
#define NHEAD   16
#define DHEAD   128
#define TANH_CAP 50.0f

__device__ __forceinline__ v8f zero8() {
  v8f v;
#pragma unroll
  for (int i = 0; i < 8; ++i) v[i] = 0.0f;
  return v;
}

// Assemble a 16-half fragment from two contiguous 8-half (16B) chunks.
__device__ __forceinline__ v16h ld_frag(const _Float16* p0, const _Float16* p1) {
  union { v16h v; v8h h[2]; } u;
  u.h[0] = *(const v8h*)p0;
  u.h[1] = *(const v8h*)p1;
  return u.v;
}

__device__ __forceinline__ v8f wmma32(v16h a, v16h b, v8f c) {
  // D = A(16x32 f16) * B(32x16 f16) + C(16x16 f32)
  return __builtin_amdgcn_wmma_f32_16x16x32_f16(
      /*neg_a=*/false, a, /*neg_b=*/false, b,
      /*c_mod=*/(short)0, c, /*reuse_a=*/false, /*reuse_b=*/false);
}

// CDNA5 async DMA: copy 16 bytes global -> LDS, tracked by ASYNCcnt.
// VDST = VGPR with LDS byte address, VADDR = 64-bit global address.
__device__ __forceinline__ void async_cp_b128(void* lds, const void* g) {
  unsigned int loff = (unsigned int)(unsigned long long)(uintptr_t)lds;
  unsigned long long ga = (unsigned long long)(uintptr_t)g;
  asm volatile("global_load_async_to_lds_b128 %0, %1, off"
               :: "v"(loff), "v"(ga) : "memory");
}
__device__ __forceinline__ void wait_async0() {
  asm volatile("s_wait_asynccnt 0x0" ::: "memory");
}

// ---------------------------------------------------------------------------
// Generic C[M][N] = A[M][K] * B[N][K]^T  (B supplied row-major as [N][K] f32).
// Software-pipelined: global loads for tile k+1 are issued into VGPRs while
// the 8 WMMAs of tile k run; fp32->f16 conversion happens at LDS-store time.
// Workgroup 256 threads (8 waves), 128x128 tile, K-step 32; each wave owns a
// 32x64 register tile (2x4 accumulators) -> 8 v_wmma per K-step per wave.
// ---------------------------------------------------------------------------
template <typename TA, typename TC>
__global__ __launch_bounds__(256) void gemm_bt_wmma(
    const TA* __restrict__ A, const float* __restrict__ Bw,
    TC* __restrict__ C, int M, int N, int K)
{
  __shared__ alignas(16) _Float16 lsA[128][40];
  __shared__ alignas(16) _Float16 lsB[128][40];

  const int tid  = threadIdx.x;
  const int lane = tid & 31;
  const int wave = tid >> 5;
  const int wm   = wave >> 1;     // 0..3
  const int wn   = wave & 1;      // 0..1
  const int g    = lane >> 4;     // half-wave group
  const int c    = lane & 15;
  const int mbase = blockIdx.y * 128;
  const int nbase = blockIdx.x * 128;

  v8f acc[2][4];
#pragma unroll
  for (int am = 0; am < 2; ++am)
#pragma unroll
    for (int bn = 0; bn < 4; ++bn) acc[am][bn] = zero8();

  float aR[16], bR[16];
  auto stage_ld = [&](int kt) {
#pragma unroll
    for (int i = 0; i < 16; ++i) {
      int e  = tid + i * 256;
      int r  = e >> 5;
      int cc = e & 31;
      aR[i] = (float)A[(size_t)(mbase + r) * K + kt + cc];
      bR[i] = (float)Bw[(size_t)(nbase + r) * K + kt + cc];
    }
  };

  stage_ld(0);
  for (int kt = 0; kt < K; kt += 32) {
    // Commit staged tile (fp32 regs -> f16 LDS).
#pragma unroll
    for (int i = 0; i < 16; ++i) {
      int e  = tid + i * 256;
      int r  = e >> 5;
      int cc = e & 31;
      lsA[r][cc] = (_Float16)aR[i];
      lsB[r][cc] = (_Float16)bR[i];
    }
    __syncthreads();

    // Kick off global loads for the next tile (overlap with WMMAs below).
    if (kt + 32 < K) stage_ld(kt + 32);

    v16h af[2], bf[4];
#pragma unroll
    for (int am = 0; am < 2; ++am) {
      const _Float16* p = &lsA[wm * 32 + am * 16 + c][g * 8];
      af[am] = ld_frag(p, p + 16);          // K chunks 8g and 16+8g
    }
#pragma unroll
    for (int bn = 0; bn < 4; ++bn) {
      const _Float16* p = &lsB[wn * 64 + bn * 16 + c][g * 8];
      bf[bn] = ld_frag(p, p + 16);          // K chunks 8g and 16+8g
    }
#pragma unroll
    for (int am = 0; am < 2; ++am)
#pragma unroll
      for (int bn = 0; bn < 4; ++bn)
        acc[am][bn] = wmma32(af[am], bf[bn], acc[am][bn]);
    __syncthreads();
  }

  // Epilogue: row M = base + r + 8*g, col N = base + (lane&15).
#pragma unroll
  for (int am = 0; am < 2; ++am)
#pragma unroll
    for (int bn = 0; bn < 4; ++bn)
#pragma unroll
      for (int r = 0; r < 8; ++r) {
        int row = mbase + wm * 32 + am * 16 + r + 8 * g;
        int col = nbase + wn * 64 + bn * 16 + c;
        C[(size_t)row * N + col] = (TC)acc[am][bn][r];
      }
}

// ---------------------------------------------------------------------------
// RoPE + layout shuffle.
//   Qh[h][s][dh] = rope(Q)[s][h,dh] * 1/sqrt(DH)   (scale folded into Q)
//   Kh[h][s][dh] = rope(K)[s][h,dh]
//   Vt[h][dh][s] = V[s][h,dh]                       (transposed for PV B-frags)
// ---------------------------------------------------------------------------
__global__ __launch_bounds__(256) void rope_split_kernel(
    const _Float16* __restrict__ Q, const _Float16* __restrict__ Kin,
    const _Float16* __restrict__ V, const float* __restrict__ cosb,
    const float* __restrict__ sinb, _Float16* __restrict__ Qh,
    _Float16* __restrict__ Kh, _Float16* __restrict__ Vt)
{
  int idx = blockIdx.x * 256 + threadIdx.x;            // over S*H*DH
  int dh  = idx & (DHEAD - 1);
  int t   = idx >> 7;
  int h   = t & (NHEAD - 1);
  int s   = t >> 4;
  int p   = h * DHEAD + dh;

  float co = cosb[s * DHEAD + dh];
  float si = sinb[s * DHEAD + dh];
  int   pr  = (dh < 64) ? (p + 64) : (p - 64);
  float sgn = (dh < 64) ? -1.0f : 1.0f;

  size_t src  = (size_t)s * DMODEL + p;
  size_t srcr = (size_t)s * DMODEL + pr;

  float q  = (float)Q[src],   qr = sgn * (float)Q[srcr];
  float k  = (float)Kin[src], kr = sgn * (float)Kin[srcr];

  const float qscale = 0.08838834764831845f;           // 1/sqrt(128)
  Qh[((size_t)h * S_LEN + s) * DHEAD + dh] = (_Float16)((q * co + qr * si) * qscale);
  Kh[((size_t)h * S_LEN + s) * DHEAD + dh] = (_Float16)(k * co + kr * si);
  Vt[((size_t)h * DHEAD + dh) * S_LEN + s] = V[src];
}

// ---------------------------------------------------------------------------
// Flash attention with tanh soft-cap + causal mask.
// Grid (S/64, H), 128 threads = 4 waves; each wave owns 16 query rows.
// K-tile (32x128) and V-tile (128x32) are staged into LDS once per block via
// async-to-LDS DMA, double buffered (DMA for block i+1 overlaps block i's
// 16 WMMAs + softmax). Rows padded (136 / 40 halves) for bank spread.
// ---------------------------------------------------------------------------
__global__ __launch_bounds__(128) void flash_attn_kernel(
    const _Float16* __restrict__ Qh, const _Float16* __restrict__ Kh,
    const _Float16* __restrict__ Vt, _Float16* __restrict__ O)
{
  __shared__ alignas(16) _Float16 Kt[2][32][136];   // [key][dh], padded
  __shared__ alignas(16) _Float16 Vl[2][128][40];   // [dh][key], padded
  __shared__ alignas(16) _Float16 Plds[4][16][40];  // per-wave P relayout

  const int h    = blockIdx.y;
  const int lane = threadIdx.x & 31;
  const int wave = threadIdx.x >> 5;
  const int g    = lane >> 4;
  const int c    = lane & 15;
  const int q0   = blockIdx.x * 64 + wave * 16;

  _Float16 (*P)[40] = Plds[wave];

  // Async-stage one 32-key block of K ([32][128], contiguous 8KB in global)
  // and V ([128][32] slice of Vt with row stride S) into LDS buffer b.
  const _Float16* KhB = Kh + (size_t)h * S_LEN * DHEAD;
  const _Float16* VtB = Vt + (size_t)h * DHEAD * S_LEN;
  auto stage_kv = [&](int b, int kb) {
    const _Float16* gk = KhB + (size_t)kb * DHEAD;
#pragma unroll
    for (int i = 0; i < 2; ++i) {                 // 32 rows x 8 chunks / 128 thr
      int idx = threadIdx.x + i * 128;
      int row = idx >> 3, ch = idx & 7;
      async_cp_b128(&Kt[b][row][ch * 8], gk + row * DHEAD + ch * 8);
    }
#pragma unroll
    for (int i = 0; i < 4; ++i) {                 // 128 rows x 4 chunks / 128 thr
      int idx = threadIdx.x + i * 128;
      int row = idx >> 2, ch = idx & 3;
      async_cp_b128(&Vl[b][row][ch * 8], VtB + (size_t)row * S_LEN + kb + ch * 8);
    }
  };

  // Q fragments (16 rows x 128 dh) held in registers: 4 A-frags of 16x32.
  const _Float16* Qb = Qh + ((size_t)h * S_LEN + q0 + c) * DHEAD;
  v16h qf[4];
#pragma unroll
  for (int t = 0; t < 4; ++t) {
    const _Float16* p = Qb + t * 32 + g * 8;
    qf[t] = ld_frag(p, p + 16);
  }

  v8f o[8];
#pragma unroll
  for (int nb = 0; nb < 8; ++nb) o[nb] = zero8();
  float m[8], l[8];
#pragma unroll
  for (int r = 0; r < 8; ++r) { m[r] = -1.0e30f; l[r] = 0.0f; }

  const int nblk = (blockIdx.x * 64 + 64) >> 5;        // causal upper bound / 32
  int buf = 0;
  stage_kv(0, 0);

  for (int ib = 0; ib < nblk; ++ib) {
    const int kb = ib * 32;
    wait_async0();
    __syncthreads();                                   // K/V tile visible to all
    if (ib + 1 < nblk) stage_kv(buf ^ 1, kb + 32);     // overlap next DMA

    // ---- scores: S = Q * K^T  (two 16x16 tiles over 32 keys) ----
    v8f s0 = zero8(), s1 = zero8();
#pragma unroll
    for (int t = 0; t < 4; ++t) {
      const _Float16* k0 = &Kt[buf][c][t * 32 + g * 8];
      const _Float16* k1 = &Kt[buf][16 + c][t * 32 + g * 8];
      s0 = wmma32(qf[t], ld_frag(k0, k0 + 16), s0);
      s1 = wmma32(qf[t], ld_frag(k1, k1 + 16), s1);
    }

    // ---- softcap + causal mask + online softmax (C-layout) ----
    float corrv[8];
#pragma unroll
    for (int r = 0; r < 8; ++r) {
      int q = q0 + r + 8 * g;
      float a = s0[r];
      a = TANH_CAP * tanhf(a * (1.0f / TANH_CAP));
      if (kb + c > q) a = -1.0e30f;
      float b = s1[r];
      b = TANH_CAP * tanhf(b * (1.0f / TANH_CAP));
      if (kb + 16 + c > q) b = -1.0e30f;

      float mx = fmaxf(a, b);
#pragma unroll
      for (int off = 1; off < 16; off <<= 1)
        mx = fmaxf(mx, __shfl_xor(mx, off, 32));       // stays within 16-lane half
      float mnew = fmaxf(m[r], mx);
      float pa = __expf(a - mnew);
      float pb = __expf(b - mnew);
      float rs = pa + pb;
#pragma unroll
      for (int off = 1; off < 16; off <<= 1)
        rs += __shfl_xor(rs, off, 32);
      float corr = __expf(m[r] - mnew);
      l[r] = l[r] * corr + rs;
      m[r] = mnew;
      corrv[r] = corr;
      s0[r] = pa;
      s1[r] = pb;
    }
#pragma unroll
    for (int nb = 0; nb < 8; ++nb)
#pragma unroll
      for (int r = 0; r < 8; ++r) o[nb][r] *= corrv[r];

    // ---- relayout P (C-layout) -> A-fragment via LDS ----
    __syncthreads();
#pragma unroll
    for (int r = 0; r < 8; ++r) {
      P[r + 8 * g][c]      = (_Float16)s0[r];
      P[r + 8 * g][16 + c] = (_Float16)s1[r];
    }
    __syncthreads();
    const _Float16* pp = &P[c][0];
    v16h pf = ld_frag(pp + g * 8, pp + 16 + g * 8);

    // ---- O += P * V  (B-frags from LDS V tile) ----
#pragma unroll
    for (int nb = 0; nb < 8; ++nb) {
      const _Float16* vb = &Vl[buf][nb * 16 + c][g * 8];
      o[nb] = wmma32(pf, ld_frag(vb, vb + 16), o[nb]);
    }
    buf ^= 1;
  }

  // ---- normalize and write O[s][h*DH + dh] (f16) ----
#pragma unroll
  for (int nb = 0; nb < 8; ++nb)
#pragma unroll
    for (int r = 0; r < 8; ++r) {
      int q   = q0 + r + 8 * g;
      int col = h * DHEAD + nb * 16 + c;
      O[(size_t)q * DMODEL + col] = (_Float16)(o[nb][r] / l[r]);
    }
}

// ---------------------------------------------------------------------------
// Launch: QKV GEMMs -> RoPE/layout -> flash attention -> output projection.
// Workspace (48 MB): [Qf16][Kf16][Vf16][Qh][Kh][Vt]; attn output reuses Qf16.
// ---------------------------------------------------------------------------
extern "C" void kernel_launch(void* const* d_in, const int* in_sizes, int n_in,
                              void* d_out, int out_size, void* d_ws, size_t ws_size,
                              hipStream_t stream) {
  const float* x    = (const float*)d_in[0];
  const float* cosb = (const float*)d_in[1];
  const float* sinb = (const float*)d_in[2];
  // d_in[3] = dense mask: unused (causal mask applied analytically)
  const float* wq = (const float*)d_in[4];
  const float* wk = (const float*)d_in[5];
  const float* wv = (const float*)d_in[6];
  const float* wo = (const float*)d_in[7];
  float* out = (float*)d_out;

  char* ws = (char*)d_ws;
  const size_t MAT = (size_t)S_LEN * DMODEL * sizeof(_Float16);   // 8 MB
  _Float16* Qf = (_Float16*)(ws + 0 * MAT);
  _Float16* Kf = (_Float16*)(ws + 1 * MAT);
  _Float16* Vf = (_Float16*)(ws + 2 * MAT);
  _Float16* Qh = (_Float16*)(ws + 3 * MAT);
  _Float16* Kh = (_Float16*)(ws + 4 * MAT);
  _Float16* Vt = (_Float16*)(ws + 5 * MAT);
  _Float16* AO = Qf;   // Q f16 is dead after rope_split; reuse for attn output

  dim3 gemm_grid(DMODEL / 128, S_LEN / 128);
  gemm_bt_wmma<float, _Float16><<<gemm_grid, 256, 0, stream>>>(
      x, wq, Qf, S_LEN, DMODEL, DMODEL);
  gemm_bt_wmma<float, _Float16><<<gemm_grid, 256, 0, stream>>>(
      x, wk, Kf, S_LEN, DMODEL, DMODEL);
  gemm_bt_wmma<float, _Float16><<<gemm_grid, 256, 0, stream>>>(
      x, wv, Vf, S_LEN, DMODEL, DMODEL);

  rope_split_kernel<<<(S_LEN * NHEAD * DHEAD) / 256, 256, 0, stream>>>(
      Qf, Kf, Vf, cosb, sinb, Qh, Kh, Vt);

  flash_attn_kernel<<<dim3(S_LEN / 64, NHEAD), 128, 0, stream>>>(
      Qh, Kh, Vt, AO);

  gemm_bt_wmma<_Float16, float><<<gemm_grid, 256, 0, stream>>>(
      AO, wo, out, S_LEN, DMODEL, DMODEL);
}
